// GraphormerLinearAttention_74844100100348
// MI455X (gfx1250) — matrix operand order repaired and checked by reading, not compiled, not run on hardware
//
#include <hip/hip_runtime.h>
#include <hip/hip_bf16.h>

// ---------------------------------------------------------------------------
// Graphormer linear attention, MI455X (gfx1250, wave32).
// Big GEMMs (x@Wq/Wk/Wv, attn@Wo) via v_wmma_f32_16x16x32_bf16 (f32 accum).
// prior-mean bias pass is the HBM-bound term (256 MB); everything else is
// small or L2-resident.
// ---------------------------------------------------------------------------

typedef __attribute__((ext_vector_type(16))) __bf16 v16bf;
typedef __attribute__((ext_vector_type(8)))  __bf16 v8bf;
typedef __attribute__((ext_vector_type(8)))  float  v8f;

#define B_   4
#define N_   4096
#define D_   256
#define H_   8
#define HD_  32
#define ROWS (B_ * N_)   // 16384

__device__ __forceinline__ float elu1(float x) {
  // elu(x)+1 : x>0 ? x+1 : exp(x)
  return x > 0.f ? x + 1.f : __expf(x);
}

// ---------------- Pass 1: bias[b,n] = mean(prior[b,n,:]) -------------------
// One wave per row, float4 streaming; this pass is the bandwidth roofline.
__global__ __launch_bounds__(256) void k_bias(const float* __restrict__ prior,
                                              float* __restrict__ bias) {
  int wave = blockIdx.x * 8 + (threadIdx.x >> 5);
  int lane = threadIdx.x & 31;
  const float4* row = (const float4*)(prior + (size_t)wave * N_);
  float s = 0.f;
  for (int i = lane; i < N_ / 4; i += 32) {
    __builtin_prefetch(&row[i + 128], 0, 0);   // global_prefetch_b8, speculative
    float4 f = row[i];
    s += (f.x + f.y) + (f.z + f.w);
  }
#pragma unroll
  for (int off = 16; off > 0; off >>= 1) s += __shfl_xor(s, off, 32);
  if (lane == 0) bias[wave] = s * (1.0f / (float)N_);
}

// ---------------- Pass 2: weights f32 -> bf16 ------------------------------
__global__ __launch_bounds__(256) void k_cvt(const float* __restrict__ src,
                                             __bf16* __restrict__ dst, int n) {
  int i = blockIdx.x * blockDim.x + threadIdx.x;
  if (i < n) dst[i] = (__bf16)src[i];
}

// ---------------- WMMA fragment helpers ------------------------------------
// A (16x32 bf16, M x K) per ISA: lane l -> m=l&15, half=l>>4;
// elements 0..7  cover K = kc + half*8 + 0..7
// elements 8..15 cover K = kc + 16 + half*8 + 0..7
__device__ __forceinline__ void load_a_f32(const float* __restrict__ X,
                                           int row0, int lane, v16bf a[8]) {
  int m = lane & 15, half = lane >> 4;
  const float* xr = X + (size_t)(row0 + m) * D_;
#pragma unroll
  for (int kc = 0; kc < 8; ++kc) {
    int kb = kc * 32;
    float4 f0 = *(const float4*)(xr + kb + half * 8);
    float4 f1 = *(const float4*)(xr + kb + half * 8 + 4);
    float4 f2 = *(const float4*)(xr + kb + 16 + half * 8);
    float4 f3 = *(const float4*)(xr + kb + 16 + half * 8 + 4);
    v16bf t;
    t[0]=(__bf16)f0.x; t[1]=(__bf16)f0.y; t[2]=(__bf16)f0.z; t[3]=(__bf16)f0.w;
    t[4]=(__bf16)f1.x; t[5]=(__bf16)f1.y; t[6]=(__bf16)f1.z; t[7]=(__bf16)f1.w;
    t[8]=(__bf16)f2.x; t[9]=(__bf16)f2.y; t[10]=(__bf16)f2.z; t[11]=(__bf16)f2.w;
    t[12]=(__bf16)f3.x; t[13]=(__bf16)f3.y; t[14]=(__bf16)f3.z; t[15]=(__bf16)f3.w;
    a[kc] = t;
  }
}

__device__ __forceinline__ void load_a_bf16(const __bf16* __restrict__ X,
                                            int row0, int lane, v16bf a[8]) {
  int m = lane & 15, half = lane >> 4;
  const __bf16* xr = X + (size_t)(row0 + m) * D_;
#pragma unroll
  for (int kc = 0; kc < 8; ++kc) {
    int kb = kc * 32;
    v8bf lo = *(const v8bf*)(xr + kb + half * 8);
    v8bf hi = *(const v8bf*)(xr + kb + 16 + half * 8);
    a[kc] = __builtin_shufflevector(lo, hi, 0,1,2,3,4,5,6,7,8,9,10,11,12,13,14,15);
  }
}

// B (32x16 bf16, K x N): lane l -> n=l&15, half=l>>4; same K order as A:
// elements 0..7 cover K = half*8 + 0..7, elements 8..15 cover K = 16 + half*8 + 0..7.
// B element (k,n) = W[n][k]  (out = X @ W^T), so a lane reads two 8-element
// runs of W's row n.
__device__ __forceinline__ void gemm_tile16(const v16bf a[8],
                                            const __bf16* __restrict__ W,
                                            int n0, int lane, v8f& c) {
  int half = lane >> 4;
  const __bf16* wrow = W + (size_t)(n0 + (lane & 15)) * D_;
#pragma unroll
  for (int kc = 0; kc < 8; ++kc) {
    v8bf blo = *(const v8bf*)(wrow + kc * 32 + half * 8);
    v8bf bhi = *(const v8bf*)(wrow + kc * 32 + 16 + half * 8);
    const v16bf b = __builtin_shufflevector(blo, bhi, 0,1,2,3,4,5,6,7,8,9,10,11,12,13,14,15);
    c = __builtin_amdgcn_wmma_f32_16x16x32_bf16(false, a[kc], false, b,
                                                (short)0, c, false, false);
  }
}

// ---------------- Pass 3: fused q/k/v projection + feature maps ------------
// One wave per 16-row tile; A fragments for the whole K=256 stay in registers
// and are reused across all 3 weights x 16 column tiles (x read exactly once).
__global__ __launch_bounds__(128) void k_qkv(const float* __restrict__ x,
                                             const float* __restrict__ bias,
                                             const __bf16* __restrict__ Wq,
                                             const __bf16* __restrict__ Wk,
                                             const __bf16* __restrict__ Wv,
                                             const float* __restrict__ bv,
                                             __bf16* __restrict__ q16,
                                             __bf16* __restrict__ k16,
                                             __bf16* __restrict__ v16) {
  int wave = blockIdx.x * 4 + (threadIdx.x >> 5);
  int lane = threadIdx.x & 31;
  int row0 = wave * 16;
  int half = lane >> 4;

  v16bf a[8];
  load_a_f32(x, row0, lane, a);

  float biasr[8];
#pragma unroll
  for (int r = 0; r < 8; ++r) biasr[r] = bias[row0 + r + 8 * half];

  for (int p = 0; p < 3; ++p) {
    const __bf16* W = (p == 0) ? Wq : (p == 1) ? Wk : Wv;
    __bf16* dst = (p == 0) ? q16 : (p == 1) ? k16 : v16;
    for (int nt = 0; nt < 16; ++nt) {
      int n0 = nt * 16;
      v8f c = {};
      gemm_tile16(a, W, n0, lane, c);
      int n = n0 + (lane & 15);
      float addv = (p == 2) ? bv[n] : 0.f;
#pragma unroll
      for (int r = 0; r < 8; ++r) {
        int row = row0 + r + 8 * half;       // ISA C/D layout: M = r + 8*half
        float val = c[r];
        if (p == 0) val += biasr[r];
        if (p == 2) val += addv;
        dst[(size_t)row * D_ + n] = (__bf16)elu1(val);
      }
    }
  }
}

// ---------------- Pass 4: kv[b,h,m,d] = sum_n k[.,d]*v[.,m]; ksum ----------
// One workgroup per (b,h); 4096-deep reduction staged through LDS.
__global__ __launch_bounds__(256) void k_kvstate(const __bf16* __restrict__ k16,
                                                 const __bf16* __restrict__ v16,
                                                 float* __restrict__ kv,
                                                 float* __restrict__ ksum) {
  int bh = blockIdx.x;              // 0..31
  int b = bh >> 3, h = bh & 7;
  __shared__ __align__(16) __bf16 ks[64][32];
  __shared__ __align__(16) __bf16 vs[64][32];
  int t = threadIdx.x;
  int m  = t >> 3;                  // 0..31
  int d0 = (t & 7) * 4;             // 4 consecutive d's per thread
  int lr = t >> 2;                  // load row within chunk
  int lc = (t & 3) * 8;             // load col (8 bf16 = one v8bf)
  float acc[4] = {0.f, 0.f, 0.f, 0.f};
  float sk = 0.f;

  for (int c0 = 0; c0 < N_; c0 += 64) {
    size_t base = ((size_t)(b * N_ + c0 + lr)) * D_ + h * HD_ + lc;
    *(v8bf*)&ks[lr][lc] = *(const v8bf*)(k16 + base);
    *(v8bf*)&vs[lr][lc] = *(const v8bf*)(v16 + base);
    __syncthreads();
#pragma unroll 4
    for (int r = 0; r < 64; ++r) {
      float vv = (float)vs[r][m];
#pragma unroll
      for (int j = 0; j < 4; ++j) acc[j] += vv * (float)ks[r][d0 + j];
    }
    if (t < 32) {
      for (int r = 0; r < 64; ++r) sk += (float)ks[r][t];
    }
    __syncthreads();
  }
  float* kvp = kv + ((size_t)bh * HD_ + m) * HD_ + d0;
#pragma unroll
  for (int j = 0; j < 4; ++j) kvp[j] = acc[j];
  if (t < 32) ksum[bh * HD_ + t] = sk;
}

// ---------------- Pass 5: attn rows (normalizer z, mask, q . kv) -----------
// One wave per row; lane = output dim m within each head.
__global__ __launch_bounds__(256) void k_attn(const __bf16* __restrict__ q16,
                                              const float* __restrict__ kv,
                                              const float* __restrict__ ksum,
                                              __bf16* __restrict__ attn16) {
  int wv = threadIdx.x >> 5, lane = threadIdx.x & 31;
  int row = blockIdx.x * 8 + wv;
  int b = row >> 12;                // row / N_
  __shared__ __align__(16) __bf16 qs[8][D_];
  *(v8bf*)&qs[wv][lane * 8] = *(const v8bf*)(q16 + (size_t)row * D_ + lane * 8);
  __syncthreads();

  for (int h = 0; h < H_; ++h) {
    const float* kvr = kv + ((size_t)(b * H_ + h) * HD_ + lane) * HD_;
    const float* ksr = ksum + (b * H_ + h) * HD_;
    float s = 0.f, acc = 0.f;
    bool nz = false;
#pragma unroll 8
    for (int d = 0; d < HD_; ++d) {
      float qd = (float)qs[wv][h * HD_ + d];
      s   += qd * ksr[d];
      acc += qd * kvr[d];
      nz = nz || (qd != 0.f);
    }
    float z = 1.0f / (s + 1e-9f);
    float o = nz ? acc * z : 0.f;   // padding mask: all-q==0 -> 0
    attn16[(size_t)row * D_ + h * HD_ + lane] = (__bf16)o;
  }
}

// ---------------- Pass 6: out = attn @ Wo^T + bo (f32 output) --------------
__global__ __launch_bounds__(128) void k_out(const __bf16* __restrict__ attn16,
                                             const __bf16* __restrict__ Wo,
                                             const float* __restrict__ bo,
                                             float* __restrict__ out) {
  int wave = blockIdx.x * 4 + (threadIdx.x >> 5);
  int lane = threadIdx.x & 31;
  int row0 = wave * 16;
  int half = lane >> 4;

  v16bf a[8];
  load_a_bf16(attn16, row0, lane, a);

  for (int nt = 0; nt < 16; ++nt) {
    int n0 = nt * 16;
    v8f c = {};
    gemm_tile16(a, Wo, n0, lane, c);
    int n = n0 + (lane & 15);
    float bov = bo[n];
#pragma unroll
    for (int r = 0; r < 8; ++r) {
      int row = row0 + r + 8 * half;
      out[(size_t)row * D_ + n] = c[r] + bov;
    }
  }
}

// ---------------------------------------------------------------------------
extern "C" void kernel_launch(void* const* d_in, const int* in_sizes, int n_in,
                              void* d_out, int out_size, void* d_ws, size_t ws_size,
                              hipStream_t stream) {
  const float* x     = (const float*)d_in[0];
  const float* prior = (const float*)d_in[1];
  const float* Wq    = (const float*)d_in[2];
  const float* Wk    = (const float*)d_in[3];
  const float* Wv    = (const float*)d_in[4];
  const float* bv    = (const float*)d_in[5];
  const float* Wo    = (const float*)d_in[6];
  const float* bo    = (const float*)d_in[7];
  float* out = (float*)d_out;

  // Workspace layout (~26 MB used):
  char* ws = (char*)d_ws;
  float*  bias = (float*)(ws + 0);                      //  64 KB
  __bf16* Wq16 = (__bf16*)(ws + (1u << 20));            // 128 KB each
  __bf16* Wk16 = Wq16 + 65536;
  __bf16* Wv16 = Wk16 + 65536;
  __bf16* Wo16 = Wv16 + 65536;
  float*  kvst = (float*)(ws + (1u << 20) + (512u << 10)); // 128 KB
  float*  ksum = (float*)(ws + (1u << 20) + (640u << 10)); //   4 KB
  __bf16* q16  = (__bf16*)(ws + (2u << 20));            // 8 MB
  __bf16* k16  = q16 + (size_t)ROWS * D_;               // 8 MB
  __bf16* v16  = k16 + (size_t)ROWS * D_;               // 8 MB
  __bf16* attn16 = v16;  // reuse v-buffer: kv pass has consumed v before attn

  // 1) bias = mean(prior, -1)                 [HBM-bound, 256 MB]
  k_bias<<<ROWS / 8, 256, 0, stream>>>(prior, bias);
  // 2) weights -> bf16 (L2-hot afterwards)
  k_cvt<<<256, 256, 0, stream>>>(Wq, Wq16, D_ * D_);
  k_cvt<<<256, 256, 0, stream>>>(Wk, Wk16, D_ * D_);
  k_cvt<<<256, 256, 0, stream>>>(Wv, Wv16, D_ * D_);
  k_cvt<<<256, 256, 0, stream>>>(Wo, Wo16, D_ * D_);
  // 3) fused q/k/v projections + feature maps [WMMA bf16]
  k_qkv<<<ROWS / 16 / 4, 128, 0, stream>>>(x, bias, Wq16, Wk16, Wv16, bv,
                                           q16, k16, v16);
  // 4) kv state + ksum per (b,h)
  k_kvstate<<<B_ * H_, 256, 0, stream>>>(k16, v16, kvst, ksum);
  // 5) attention rows (z, mask, q.kv)
  k_attn<<<ROWS / 8, 256, 0, stream>>>(q16, kvst, ksum, attn16);
  // 6) output projection + bo                 [WMMA bf16]
  k_out<<<ROWS / 16 / 4, 128, 0, stream>>>(attn16, Wo16, bo, out);

  (void)in_sizes; (void)n_in; (void)out_size; (void)ws_size;
}
